// MambaMixer_10574209482909
// MI455X (gfx1250) — matrix operand order repaired and checked by reading, not compiled, not run on hardware
//
#include <hip/hip_runtime.h>
#include <hip/hip_bf16.h>

// ---------------------------------------------------------------------------
// Bidirectional Mamba (Vision-Mamba) mixer for MI455X (gfx1250, wave32).
// GEMMs run on v_wmma_f32_16x16x32_bf16; scan is a lane-per-(b,d,n) recurrence
// with B/C staged into LDS via global_load_async_to_lds_b128 (ASYNCcnt path)
// and 16-lane shuffle reductions.
// ---------------------------------------------------------------------------

#define BATCH   8
#define LSEQ    1024
#define DMODEL  768
#define DINNER  1536
#define DSTATE  16
#define DTRANK  48
#define RPAD    64      // dt-rank padded to a multiple of 32 (K for WMMA)
#define NE      80      // DT_RANK + 2*D_STATE
#define NXZ     3072    // 2*D_INNER
#define MROWS   (BATCH * LSEQ)   // 8192
#define SPITCH  68      // LDS row pitch for scan staging (bank-conflict free)

typedef __attribute__((ext_vector_type(16))) __bf16 v16bf;
typedef __attribute__((ext_vector_type(8)))  __bf16 v8bf;
typedef __attribute__((ext_vector_type(8)))  float  v8f;

__device__ __forceinline__ unsigned short f2bf(float x) {
    union { __hip_bfloat16 h; unsigned short s; } u;
    u.h = __float2bfloat16(x);
    return u.s;
}

// Low 32 bits of a generic pointer to LDS == byte offset within the
// workgroup's LDS allocation (flat LDS address = {aperture, offset}).
__device__ __forceinline__ unsigned lds_off(const void* p) {
    return (unsigned)(unsigned long long)(uintptr_t)p;
}

// ---- WMMA fragment loaders (layouts per CDNA5 ISA 7.12.2, wave32) ----------

// A (16x32 bf16, row major, lda in elements; lda*2 must be 16B-multiple):
// lanes 0-15: M=lane, VGPR0-3 = K 0..7, VGPR4-7 = K 16..23
// lanes16-31: M=lane-16, VGPR0-3 = K 8..15, VGPR4-7 = K 24..31
__device__ __forceinline__ v16bf load_fragA(const __bf16* A, int lda,
                                            int m0, int k0, int lane) {
    const int r16 = lane & 15, hl = lane >> 4;
    const __bf16* p = A + (size_t)(m0 + r16) * lda + k0 + hl * 8;
    union { v16bf v; v8bf h[2]; } f;
    f.h[0] = *(const v8bf*)(p);        // K = k0 + hl*8 + [0..7]
    f.h[1] = *(const v8bf*)(p + 16);   // K = k0 + 16 + hl*8 + [0..7]
    return f.v;
}

// B (32x16 bf16). Weights stored (N,K) row major => column n contiguous in K.
// lanes 0-15: N=lane, K = 0..7,16..23 ; lanes16-31: N=lane-16, K = 8..15,24..31
__device__ __forceinline__ v16bf load_fragB(const __bf16* B, int ldb,
                                            int n0, int k0, int lane) {
    const int r16 = lane & 15, hl = lane >> 4;
    const __bf16* p = B + (size_t)(n0 + r16) * ldb + k0 + hl * 8;
    union { v16bf v; v8bf h[2]; } f;
    f.h[0] = *(const v8bf*)(p);
    f.h[1] = *(const v8bf*)(p + 16);
    return f.v;
}

__device__ __forceinline__ v8f wmma_bf16(v16bf a, v16bf b, v8f c) {
    return __builtin_amdgcn_wmma_f32_16x16x32_bf16(
        false, a, false, b, (short)0, c, false, false);
}

// D/C layout: element r of v8f <-> row (m0 + (lane>>4)*8 + r), col (n0 + lane&15)

// ---------------------------------------------------------------------------
// Conversion kernels
// ---------------------------------------------------------------------------
__global__ void k_cvt(const float* __restrict__ src,
                      unsigned short* __restrict__ dst, int n) {
    int i = blockIdx.x * 256 + threadIdx.x;
    if (i < n) dst[i] = f2bf(src[i]);
}

// dt_proj_w (DINNER,48) -> bf16 (DINNER,64) with zero K-padding
__global__ void k_cvt_pad_dtw(const float* __restrict__ src,
                              unsigned short* __restrict__ dst) {
    int i = blockIdx.x * 256 + threadIdx.x;
    if (i < DINNER * RPAD) {
        int d = i >> 6, r = i & 63;
        dst[i] = (r < DTRANK) ? f2bf(src[d * DTRANK + r]) : (unsigned short)0;
    }
}

__global__ void k_fill_zero16(unsigned short* __restrict__ dst, int n) {
    int i = blockIdx.x * 256 + threadIdx.x;
    if (i < n) dst[i] = 0;
}

// ---------------------------------------------------------------------------
// GEMM 1: xz[b,e,l] = sum_d X[b*L+l, d] * W1[e, d]      M=8192 N=3072 K=768
// 8 waves/block, 32x32 per wave (2x2 of 16x16), block tile 64x128.
// Store transposed into (b, e, l).
// ---------------------------------------------------------------------------
__global__ void __launch_bounds__(256)
k_gemm_inproj(const unsigned short* __restrict__ Abits,
              const unsigned short* __restrict__ Bbits,
              float* __restrict__ xz) {
    const __bf16* A = (const __bf16*)Abits;
    const __bf16* B = (const __bf16*)Bbits;
    const int lane = threadIdx.x & 31, wave = threadIdx.x >> 5;
    const int m0 = blockIdx.x * 64 + (wave >> 2) * 32;
    const int n0 = blockIdx.y * 128 + (wave & 3) * 32;
    const int K = DMODEL;
    v8f a00 = {}, a01 = {}, a10 = {}, a11 = {};
    for (int k0 = 0; k0 < K; k0 += 32) {
        if (k0 + 32 < K) {
            __builtin_prefetch(A + (size_t)(m0 + (lane & 15)) * K + k0 + 32, 0, 1);
            __builtin_prefetch(B + (size_t)(n0 + (lane & 15)) * K + k0 + 32, 0, 1);
        }
        v16bf fa0 = load_fragA(A, K, m0, k0, lane);
        v16bf fa1 = load_fragA(A, K, m0 + 16, k0, lane);
        v16bf fb0 = load_fragB(B, K, n0, k0, lane);
        v16bf fb1 = load_fragB(B, K, n0 + 16, k0, lane);
        a00 = wmma_bf16(fa0, fb0, a00);
        a01 = wmma_bf16(fa0, fb1, a01);
        a10 = wmma_bf16(fa1, fb0, a10);
        a11 = wmma_bf16(fa1, fb1, a11);
    }
    const int r16 = lane & 15, hl = lane >> 4;
    v8f acc[2][2] = {{a00, a01}, {a10, a11}};
#pragma unroll
    for (int i = 0; i < 2; ++i)
#pragma unroll
        for (int j = 0; j < 2; ++j) {
            int n = n0 + j * 16 + r16;
#pragma unroll
            for (int r = 0; r < 8; ++r) {
                int m = m0 + i * 16 + hl * 8 + r;
                int b = m >> 10, l = m & 1023;
                xz[(((size_t)b * NXZ + n) << 10) + l] = acc[i][j][r];
            }
        }
}

// ---------------------------------------------------------------------------
// Depthwise causal conv(k=4) + SiLU on x-half of xz (sequence possibly
// reversed for the backward branch). Writes fp32 (b,d,t) and bf16 (b,t,d).
// ---------------------------------------------------------------------------
__global__ void __launch_bounds__(256)
k_conv(const float* __restrict__ xz, const float* __restrict__ cw,
       const float* __restrict__ cb, float* __restrict__ xc,
       unsigned short* __restrict__ xcbf, int fwd) {
    const int t = blockIdx.x * 256 + threadIdx.x;
    const int d = blockIdx.y, b = blockIdx.z;
    const float* xrow = xz + (((size_t)b * NXZ + d) << 10);
    float acc = cb[d];
#pragma unroll
    for (int j = 0; j < 4; ++j) {
        int s = t - 3 + j;
        if (s >= 0) acc += cw[d * 4 + j] * xrow[fwd ? s : (LSEQ - 1 - s)];
    }
    float v = acc / (1.0f + __expf(-acc));   // SiLU
    xc[(((size_t)b * DINNER + d) << 10) + t] = v;
    xcbf[((size_t)((b << 10) + t)) * DINNER + d] = f2bf(v);
}

// ---------------------------------------------------------------------------
// GEMM 2: x_dbl[b,e,t] = sum_d xc[b,t,d] * x_proj_w[e,d]  M=8192 N=80 K=1536
// 5 waves/block, one 16x16 tile each. dt slice (e<48) also stored bf16 (b,t,r).
// ---------------------------------------------------------------------------
__global__ void __launch_bounds__(160)
k_gemm_xdbl(const unsigned short* __restrict__ Abits,
            const unsigned short* __restrict__ Bbits,
            float* __restrict__ xdbl, unsigned short* __restrict__ dtbf) {
    const __bf16* A = (const __bf16*)Abits;
    const __bf16* B = (const __bf16*)Bbits;
    const int lane = threadIdx.x & 31, wave = threadIdx.x >> 5;  // wave 0..4
    const int m0 = blockIdx.x * 16;
    const int n0 = wave * 16;
    v8f acc = {};
    for (int k0 = 0; k0 < DINNER; k0 += 32) {
        if (k0 + 32 < DINNER)
            __builtin_prefetch(A + (size_t)(m0 + (lane & 15)) * DINNER + k0 + 32, 0, 1);
        acc = wmma_bf16(load_fragA(A, DINNER, m0, k0, lane),
                        load_fragB(B, DINNER, n0, k0, lane), acc);
    }
    const int r16 = lane & 15, hl = lane >> 4;
    const int n = n0 + r16;
#pragma unroll
    for (int r = 0; r < 8; ++r) {
        int m = m0 + hl * 8 + r;
        int b = m >> 10, l = m & 1023;
        xdbl[(((size_t)b * NE + n) << 10) + l] = acc[r];
        if (n < DTRANK)
            dtbf[((size_t)((b << 10) + l)) * RPAD + n] = f2bf(acc[r]);
    }
}

// ---------------------------------------------------------------------------
// GEMM 3: delta[b,d,t] = softplus(sum_r dt[b,t,r]*dtw[d,r] + dtb[d])
// M=8192 N=1536 K=64 (padded). 8 waves/block, 16x16 tile each.
// ---------------------------------------------------------------------------
__global__ void __launch_bounds__(256)
k_gemm_delta(const unsigned short* __restrict__ Abits,
             const unsigned short* __restrict__ Bbits,
             const float* __restrict__ bias, float* __restrict__ delta) {
    const __bf16* A = (const __bf16*)Abits;
    const __bf16* B = (const __bf16*)Bbits;
    const int lane = threadIdx.x & 31, wave = threadIdx.x >> 5;
    const int m0 = blockIdx.x * 16;
    const int n0 = blockIdx.y * 128 + wave * 16;
    v8f acc = {};
#pragma unroll
    for (int k0 = 0; k0 < RPAD; k0 += 32)
        acc = wmma_bf16(load_fragA(A, RPAD, m0, k0, lane),
                        load_fragB(B, RPAD, n0, k0, lane), acc);
    const int r16 = lane & 15, hl = lane >> 4;
    const int n = n0 + r16;
    const float bn = bias[n];
#pragma unroll
    for (int r = 0; r < 8; ++r) {
        int m = m0 + hl * 8 + r;
        int b = m >> 10, l = m & 1023;
        float v = acc[r] + bn;
        float sp = (v > 20.0f) ? v : log1pf(__expf(v));
        delta[(((size_t)b * DINNER + n) << 10) + l] = sp;
    }
}

// ---------------------------------------------------------------------------
// Selective scan. Lane per (b,d,n): h recurrence; y reduced over 16 state
// lanes via shfl_xor(width=16). B_t/C_t staged into LDS with CDNA5 async
// global->LDS B128 DMA (ASYNCcnt), [n][t] layout, pitch 68 floats so the 16
// state lanes hit 16 distinct banks ((68 mod 64)=4 -> bank 4n+tt).
// fwd: ytmp[b,l,d] = gated y ; bwd: ycbf[b,l,d] = bf16(ytmp + gated y).
// ---------------------------------------------------------------------------
__global__ void __launch_bounds__(256)
k_scan(const float* __restrict__ delta, const float* __restrict__ u,
       const float* __restrict__ xdbl, const float* __restrict__ xz,
       const float* __restrict__ A_log, const float* __restrict__ Dp,
       float* __restrict__ ytmp, unsigned short* __restrict__ ycbf, int fwd) {
    const int tid = threadIdx.x;
    const int n = tid & 15;
    const int d = blockIdx.x * 16 + (tid >> 4);
    const int b = blockIdx.y;

    const float An = -__expf(A_log[d * DSTATE + n]);
    const float Dd = Dp[d];
    const float* drow = delta + (((size_t)b * DINNER + d) << 10);
    const float* urow = u + (((size_t)b * DINNER + d) << 10);
    const float* zrow = xz + (((size_t)b * NXZ + DINNER + d) << 10);
    const float* Bb = xdbl + (((size_t)b * NE + DTRANK) << 10);
    const float* Cb = xdbl + (((size_t)b * NE + DTRANK + DSTATE) << 10);

    __shared__ __align__(16) float Bs[DSTATE][SPITCH];
    __shared__ __align__(16) float Cs[DSTATE][SPITCH];

    // Per-thread async-DMA slot: row nn (state), 4 timesteps starting at c4.
    const int nn = tid >> 4;            // 0..15
    const int c4 = (tid & 15) * 4;      // 0,4,...,60
    const unsigned ldsB = lds_off(&Bs[nn][c4]);
    const unsigned ldsC = lds_off(&Cs[nn][c4]);
    const float* gB = Bb + ((size_t)nn << 10) + c4;
    const float* gC = Cb + ((size_t)nn << 10) + c4;

    float h = 0.0f;
    for (int t0 = 0; t0 < LSEQ; t0 += 64) {
        __syncthreads();
        // 2 x B128 per thread: 256 threads cover the full 4KB B tile + 4KB C
        asm volatile(
            "global_load_async_to_lds_b128 %0, %1, off\n\t"
            "global_load_async_to_lds_b128 %2, %3, off"
            :: "v"(ldsB), "v"(gB + t0), "v"(ldsC), "v"(gC + t0)
            : "memory");
        asm volatile("s_wait_asynccnt 0" ::: "memory");
        __syncthreads();
        for (int tt = 0; tt < 64; ++tt) {
            const int t = t0 + tt;
            float dt_ = drow[t], uu = urow[t];
            float dA = __expf(dt_ * An);
            h = dA * h + (dt_ * uu) * Bs[n][tt];
            float y = h * Cs[n][tt];
#pragma unroll
            for (int off = 8; off >= 1; off >>= 1)
                y += __shfl_xor(y, off, 16);
            if (n == 0) {
                int l = fwd ? t : (LSEQ - 1 - t);
                float z = zrow[l];
                float g = z / (1.0f + __expf(-z));        // SiLU(z)
                float val = (y + uu * Dd) * g;
                size_t oidx = ((size_t)((b << 10) + l)) * DINNER + d;
                if (fwd) ytmp[oidx] = val;
                else     ycbf[oidx] = f2bf(ytmp[oidx] + val);
            }
        }
    }
}

// ---------------------------------------------------------------------------
// GEMM 4: out[b,l,c] = sum_d y[b,l,d] * out_proj_w[c,d]  M=8192 N=768 K=1536
// Same block structure as GEMM1, row-major store (matches (B,H,W,C)).
// ---------------------------------------------------------------------------
__global__ void __launch_bounds__(256)
k_gemm_out(const unsigned short* __restrict__ Abits,
           const unsigned short* __restrict__ Bbits,
           float* __restrict__ out) {
    const __bf16* A = (const __bf16*)Abits;
    const __bf16* B = (const __bf16*)Bbits;
    const int lane = threadIdx.x & 31, wave = threadIdx.x >> 5;
    const int m0 = blockIdx.x * 64 + (wave >> 2) * 32;
    const int n0 = blockIdx.y * 128 + (wave & 3) * 32;
    const int K = DINNER;
    v8f a00 = {}, a01 = {}, a10 = {}, a11 = {};
    for (int k0 = 0; k0 < K; k0 += 32) {
        if (k0 + 32 < K) {
            __builtin_prefetch(A + (size_t)(m0 + (lane & 15)) * K + k0 + 32, 0, 1);
            __builtin_prefetch(B + (size_t)(n0 + (lane & 15)) * K + k0 + 32, 0, 1);
        }
        v16bf fa0 = load_fragA(A, K, m0, k0, lane);
        v16bf fa1 = load_fragA(A, K, m0 + 16, k0, lane);
        v16bf fb0 = load_fragB(B, K, n0, k0, lane);
        v16bf fb1 = load_fragB(B, K, n0 + 16, k0, lane);
        a00 = wmma_bf16(fa0, fb0, a00);
        a01 = wmma_bf16(fa0, fb1, a01);
        a10 = wmma_bf16(fa1, fb0, a10);
        a11 = wmma_bf16(fa1, fb1, a11);
    }
    const int r16 = lane & 15, hl = lane >> 4;
    v8f acc[2][2] = {{a00, a01}, {a10, a11}};
#pragma unroll
    for (int i = 0; i < 2; ++i)
#pragma unroll
        for (int j = 0; j < 2; ++j) {
            int n = n0 + j * 16 + r16;
#pragma unroll
            for (int r = 0; r < 8; ++r) {
                int m = m0 + i * 16 + hl * 8 + r;
                out[(size_t)m * DMODEL + n] = acc[i][j][r];
            }
        }
}

// ---------------------------------------------------------------------------
// Host-side orchestration
// ---------------------------------------------------------------------------
extern "C" void kernel_launch(void* const* d_in, const int* in_sizes, int n_in,
                              void* d_out, int out_size, void* d_ws, size_t ws_size,
                              hipStream_t stream) {
    const float* X      = (const float*)d_in[0];
    const float* W1     = (const float*)d_in[1];
    const float* cw_f   = (const float*)d_in[2];
    const float* cb_f   = (const float*)d_in[3];
    const float* xpw_f  = (const float*)d_in[4];
    const float* dtw_f  = (const float*)d_in[5];
    const float* dtb_f  = (const float*)d_in[6];
    const float* Alog_f = (const float*)d_in[7];
    const float* Dp_f   = (const float*)d_in[8];
    const float* cw_b   = (const float*)d_in[9];
    const float* cb_b   = (const float*)d_in[10];
    const float* xpw_b  = (const float*)d_in[11];
    const float* dtw_b  = (const float*)d_in[12];
    const float* dtb_b  = (const float*)d_in[13];
    const float* Alog_b = (const float*)d_in[14];
    const float* Dp_b   = (const float*)d_in[15];
    const float* OW     = (const float*)d_in[16];
    float* out = (float*)d_out;

    // bump allocator over workspace (256B aligned). Total ~315 MB.
    char* base = (char*)d_ws;
    size_t off = 0;
    auto alloc = [&](size_t bytes) -> void* {
        off = (off + 255) & ~(size_t)255;
        void* p = base + off;
        off += bytes;
        return p;
    };

    unsigned short* Xbf    = (unsigned short*)alloc((size_t)MROWS * DMODEL * 2);
    unsigned short* W1bf   = (unsigned short*)alloc((size_t)NXZ * DMODEL * 2);
    unsigned short* XPbf_f = (unsigned short*)alloc((size_t)NE * DINNER * 2);
    unsigned short* XPbf_b = (unsigned short*)alloc((size_t)NE * DINNER * 2);
    unsigned short* DTWbf_f= (unsigned short*)alloc((size_t)DINNER * RPAD * 2);
    unsigned short* DTWbf_b= (unsigned short*)alloc((size_t)DINNER * RPAD * 2);
    unsigned short* OWbf   = (unsigned short*)alloc((size_t)DMODEL * DINNER * 2);
    float* xz    = (float*)alloc((size_t)BATCH * NXZ * LSEQ * 4);
    float* xc    = (float*)alloc((size_t)BATCH * DINNER * LSEQ * 4);
    unsigned short* xcbf = (unsigned short*)alloc((size_t)MROWS * DINNER * 2);
    float* xdbl  = (float*)alloc((size_t)BATCH * NE * LSEQ * 4);
    unsigned short* dtbf = (unsigned short*)alloc((size_t)MROWS * RPAD * 2);
    float* delta = (float*)alloc((size_t)BATCH * DINNER * LSEQ * 4);
    float* ytmp  = (float*)alloc((size_t)MROWS * DINNER * 4);
    unsigned short* ycbf = (unsigned short*)alloc((size_t)MROWS * DINNER * 2);

    auto cgrid = [](int n) { return dim3((n + 255) / 256); };

    // --- bf16 conversions -------------------------------------------------
    k_cvt<<<cgrid(MROWS * DMODEL), 256, 0, stream>>>(X, Xbf, MROWS * DMODEL);
    k_cvt<<<cgrid(NXZ * DMODEL), 256, 0, stream>>>(W1, W1bf, NXZ * DMODEL);
    k_cvt<<<cgrid(NE * DINNER), 256, 0, stream>>>(xpw_f, XPbf_f, NE * DINNER);
    k_cvt<<<cgrid(NE * DINNER), 256, 0, stream>>>(xpw_b, XPbf_b, NE * DINNER);
    k_cvt_pad_dtw<<<cgrid(DINNER * RPAD), 256, 0, stream>>>(dtw_f, DTWbf_f);
    k_cvt_pad_dtw<<<cgrid(DINNER * RPAD), 256, 0, stream>>>(dtw_b, DTWbf_b);
    k_cvt<<<cgrid(DMODEL * DINNER), 256, 0, stream>>>(OW, OWbf, DMODEL * DINNER);

    // --- in_proj GEMM -----------------------------------------------------
    k_gemm_inproj<<<dim3(MROWS / 64, NXZ / 128), 256, 0, stream>>>(Xbf, W1bf, xz);

    // --- two directional branches ----------------------------------------
    for (int branch = 0; branch < 2; ++branch) {
        const int fwd = (branch == 0) ? 1 : 0;
        const float* cw   = fwd ? cw_f   : cw_b;
        const float* cb   = fwd ? cb_f   : cb_b;
        const float* dtb  = fwd ? dtb_f  : dtb_b;
        const float* Alog = fwd ? Alog_f : Alog_b;
        const float* Dpv  = fwd ? Dp_f   : Dp_b;
        unsigned short* XPbf  = fwd ? XPbf_f  : XPbf_b;
        unsigned short* DTWbf = fwd ? DTWbf_f : DTWbf_b;

        k_fill_zero16<<<cgrid(MROWS * RPAD), 256, 0, stream>>>(dtbf, MROWS * RPAD);
        k_conv<<<dim3(LSEQ / 256, DINNER, BATCH), 256, 0, stream>>>(
            xz, cw, cb, xc, xcbf, fwd);
        k_gemm_xdbl<<<dim3(MROWS / 16), 160, 0, stream>>>(xcbf, XPbf, xdbl, dtbf);
        k_gemm_delta<<<dim3(MROWS / 16, DINNER / 128), 256, 0, stream>>>(
            dtbf, DTWbf, dtb, delta);
        k_scan<<<dim3(DINNER / 16, BATCH), 256, 0, stream>>>(
            delta, xc, xdbl, xz, Alog, Dpv, ytmp, ycbf, fwd);
    }

    // --- out_proj GEMM ----------------------------------------------------
    k_gemm_out<<<dim3(MROWS / 64, DMODEL / 128), 256, 0, stream>>>(ycbf, OWbf, out);
}
